// ConvVMamba_29832842838270
// MI455X (gfx1250) — matrix operand order repaired and checked by reading, not compiled, not run on hardware
//
#include <hip/hip_runtime.h>
#include <cstdint>

// Problem constants (from the reference)
#define BB 8
#define CC 96
#define HH 64
#define WW 64
#define LL (HH*WW)        // 4096
#define KK 4
#define RR 6

typedef __attribute__((ext_vector_type(16))) _Float16 v16h;
typedef __attribute__((ext_vector_type(8)))  _Float16 v8h;
typedef __attribute__((ext_vector_type(2)))  _Float16 v2h;
typedef __attribute__((ext_vector_type(8)))  float    v8f;

__device__ __forceinline__ float gelu_f(float x){ return 0.5f*x*(1.0f+erff(x*0.70710678118654752f)); }
__device__ __forceinline__ float silu_f(float x){ return x/(1.0f+__expf(-x)); }
__device__ __forceinline__ float softplus_f(float x){ return (x>20.f)?x:log1pf(__expf(x)); }

// Cross-scan index map; involutive: reading xs_k[l] = x[g(k,l)] and
// cross_merge adds ys_k[l] at position g(k,l) as well.
__device__ __forceinline__ int scan_pos(int k, int l){
  int ll = (k & 2) ? (LL-1-l) : l;
  if (k & 1) ll = (ll & (HH-1))*WW + (ll >> 6);
  return ll;
}

// ---------------- depthwise conv (7x7 / 3x3), optional SiLU ----------------
template<int KS, bool SILU>
__global__ void dwconv_kernel(const float* __restrict__ in, const float* __restrict__ w,
                              const float* __restrict__ bias, float* __restrict__ out){
  int t = blockIdx.x*blockDim.x + threadIdx.x;
  if (t >= BB*CC*LL) return;
  int xx0 = t & (WW-1);
  int yy0 = (t >> 6) & (HH-1);
  int c   = (t / LL) % CC;
  const int P = KS/2;
  const float* wp = w + c*KS*KS;
  const float* ip = in + (t - (t & (LL-1)));   // base of this (b,c) plane
  float acc = bias[c];
  #pragma unroll
  for (int i=0;i<KS;i++){
    int yy = yy0 + i - P;
    if (yy < 0 || yy >= HH) continue;
    #pragma unroll
    for (int j=0;j<KS;j++){
      int xx = xx0 + j - P;
      if (xx < 0 || xx >= WW) continue;
      acc += ip[yy*WW + xx] * wp[i*KS + j];
    }
  }
  if (SILU) acc = silu_f(acc);
  out[t] = acc;
}

// ---------------- LayerNorm over channel dim at each (b,h,w) ----------------
__global__ void ln2d_kernel(const float* __restrict__ in, const float* __restrict__ w,
                            const float* __restrict__ b, float* __restrict__ out){
  int t = blockIdx.x*blockDim.x + threadIdx.x;
  if (t >= BB*LL) return;
  const float* ip = in  + (t >> 12)*(CC*LL) + (t & (LL-1));
  float s = 0.f, ss = 0.f;
  #pragma unroll 8
  for (int c=0;c<CC;c++){ float v = ip[c*LL]; s += v; ss += v*v; }
  float m   = s*(1.0f/CC);
  float var = ss*(1.0f/CC) - m*m;
  float inv = rsqrtf(var + 1e-5f);
  float* op = out + (t >> 12)*(CC*LL) + (t & (LL-1));
  #pragma unroll 8
  for (int c=0;c<CC;c++) op[c*LL] = (ip[c*LL]-m)*inv*w[c] + b[c];
}

// ---------------- WMMA GEMM: out[b,o,n] = act(sum_c W[o,c]*X[b,c,n] + bias[o]) (+res) ----
// grid = (LL/128, O/16, B), block = 256 (8 waves, one 16x16 tile each).
// ACT: 0 = none, 1 = exact GELU.  CIN is compile-time -> fully unrolled K loop.
template<int ACT, int CIN>
__global__ __launch_bounds__(256) void gemm_wmma_kernel(
    const float* __restrict__ X, const float* __restrict__ Wt,
    const float* __restrict__ bias, const float* __restrict__ res,
    float* __restrict__ out, int O){
  constexpr int NT = CIN/32;
  __shared__ __align__(32) _Float16 lA[2][16*32];      // A tile [m][k], shared by all waves
  __shared__ __align__(32) _Float16 lB[2][8*16*32];    // per-wave B tile, transposed [wave][n][k]
  const int lane = threadIdx.x & 31;
  const int wv   = threadIdx.x >> 5;
  const int m_base = blockIdx.y*16;
  const int n_base = blockIdx.x*128 + wv*16;
  const int nloc = lane & 15;         // A row M / B-C-D column N for this lane
  const int hf   = lane >> 4;         // lane half
  // 64-bit bases computed once; all loop offsets are 32-bit.
  const float* __restrict__ Wrow = Wt + (m_base + (threadIdx.x >> 4))*CIN + ((threadIdx.x & 15) << 1);
  const float* __restrict__ Xcol = X + (size_t)blockIdx.z*CIN*LL + n_base + nloc;
  _Float16* __restrict__ lBrow = &lB[0][(wv*16 + nloc)*32 + hf*16];

  // ---- staging helpers (phase-split so loads batch without interleaved waits) ----
  auto stageA = [&](int t, int bufi){
    // 256 threads x float2 = 512 f32 = full 16x32 tile; k is even.
    float2 w2 = *(const float2*)&Wrow[t*32];
    v2h p; p[0] = (_Float16)w2.x; p[1] = (_Float16)w2.y;
    *(v2h*)&lA[bufi][((threadIdx.x >> 4)<<5) + ((threadIdx.x & 15) << 1)] = p;
  };
  auto stageB = [&](int t, int bufi){
    float tmp[16];
    #pragma unroll
    for (int i=0;i<16;i++) tmp[i] = Xcol[(t*32 + hf*16 + i)*LL];
    v16h pk;
    #pragma unroll
    for (int i=0;i<16;i++) pk[i] = (_Float16)tmp[i];
    *(v16h*)(lBrow + bufi*(8*16*32)) = pk;            // 32B contiguous -> ds_store_b128 x2
  };

  v8f acc = {};
  stageA(0, 0);
  stageB(0, 0);
  #pragma unroll
  for (int t=0; t<NT; t++){
    __syncthreads();
    if (t+1 < NT){ stageA(t+1, (t+1)&1); stageB(t+1, (t+1)&1); }
    const int bufi = t & 1;
    // A fragment: lane holds row M = lane%16; K = 16*half + i (contiguous)
    v8h a0 = *(const v8h*)&lA[bufi][nloc*32 + hf*16];
    v8h a1 = *(const v8h*)&lA[bufi][nloc*32 + hf*16 + 8];
    v16h af;
    #pragma unroll
    for (int i=0;i<8;i++){ af[i] = a0[i]; af[8+i] = a1[i]; }
    // B fragment: lane holds col N = lane%16; K = 16*half + i (contiguous 32B)
    v16h bf = *(const v16h*)&lB[bufi][(wv*16 + nloc)*32 + hf*16];
    acc = __builtin_amdgcn_wmma_f32_16x16x32_f16(
        /*neg_a=*/false, af, /*neg_b=*/false, bf,
        /*c_mod=*/(short)0, acc, /*reuse_a=*/false, /*reuse_b=*/false);
  }
  // C/D layout: lane -> col N = lane%16; VGPR v -> row M = v + 8*half
  const size_t ooff = (size_t)blockIdx.z*O*LL;
  const int nn = n_base + nloc;
  #pragma unroll
  for (int v=0; v<8; v++){
    int o = m_base + v + 8*hf;
    float r = acc[v] + bias[o];
    if (ACT == 1) r = gelu_f(r);
    if (res) r += res[ooff + o*LL + nn];
    out[ooff + o*LL + nn] = r;
  }
}

// ---------------- x-projection: xdbl[b,k,r,l] = sum_d xpw[k,r,d] * x[b,d,g(k,l)] ----
__global__ void xdbl_kernel(const float* __restrict__ xss, const float* __restrict__ xpw,
                            float* __restrict__ xd){
  int t = blockIdx.x*blockDim.x + threadIdx.x;
  if (t >= BB*KK*LL) return;
  int l  = t & (LL-1);
  int k  = (t >> 12) & 3;
  int bi = t >> 14;
  int pos = scan_pos(k, l);
  const float* xp = xss + bi*(CC*LL) + pos;
  const float* wp = xpw + k*(RR+2)*CC;
  float acc[RR+2];
  #pragma unroll
  for (int r=0;r<RR+2;r++) acc[r] = 0.f;
  for (int d=0;d<CC;d++){
    float u = xp[d*LL];
    #pragma unroll
    for (int r=0;r<RR+2;r++) acc[r] += wp[r*CC + d]*u;
  }
  float* op = xd + (bi*KK + k)*((RR+2)*LL) + l;
  #pragma unroll
  for (int r=0;r<RR+2;r++) op[r*LL] = acc[r];
}

// ---------------- fused dt-proj + selective scan (N=1) + cross-merge --------------
__global__ void scan_kernel(const float* __restrict__ xd, const float* __restrict__ xss,
                            const float* __restrict__ dtw, const float* __restrict__ dtb,
                            const float* __restrict__ alogs, const float* __restrict__ ds,
                            float* __restrict__ ymg){
  int t = blockIdx.x*blockDim.x + threadIdx.x;
  if (t >= BB*KK*CC) return;
  int d  = t % CC;
  int k  = (t / CC) % KK;
  int bi = t / (CC*KK);
  float A     = -__expf(alogs[k*CC + d]);   // A_logs (K*C, 1)
  float Dv    = ds[k*CC + d];
  float bterm = dtb[k*CC + d];
  float w[RR];
  #pragma unroll
  for (int r=0;r<RR;r++) w[r] = dtw[(k*CC + d)*RR + r];  // dt_w (K,C,R)
  const float* xdp = xd  + (bi*KK + k)*((RR+2)*LL);
  const float* up  = xss + bi*(CC*LL) + d*LL;
  float*       yp  = ymg + bi*(CC*LL) + d*LL;
  float h = 0.f;
  for (int l=0;l<LL;l++){
    float dtv = bterm;
    #pragma unroll
    for (int r=0;r<RR;r++) dtv += w[r]*xdp[r*LL + l];
    float delta = softplus_f(dtv);
    int pos = scan_pos(k, l);
    float u  = up[pos];
    float Bv = xdp[RR*LL + l];
    float Cv = xdp[(RR+1)*LL + l];
    h = __expf(delta*A)*h + delta*u*Bv;
    atomicAdd(&yp[pos], Cv*h + Dv*u);   // cross-merge via involutive map
  }
}

// ------------------------------------------------------------------------------
extern "C" void kernel_launch(void* const* d_in, const int* in_sizes, int n_in,
                              void* d_out, int out_size, void* d_ws, size_t ws_size,
                              hipStream_t stream){
  const float* x        = (const float*)d_in[0];
  const float* cn_dw_w  = (const float*)d_in[1];
  const float* cn_dw_b  = (const float*)d_in[2];
  const float* cn_ln_w  = (const float*)d_in[3];
  const float* cn_ln_b  = (const float*)d_in[4];
  const float* cn_fc1_w = (const float*)d_in[5];
  const float* cn_fc1_b = (const float*)d_in[6];
  const float* cn_fc2_w = (const float*)d_in[7];
  const float* cn_fc2_b = (const float*)d_in[8];
  const float* v_ln1_w  = (const float*)d_in[9];
  const float* v_ln1_b  = (const float*)d_in[10];
  const float* ip_w     = (const float*)d_in[11];
  const float* ip_b     = (const float*)d_in[12];
  const float* dw_w     = (const float*)d_in[13];
  const float* dw_b     = (const float*)d_in[14];
  const float* x_proj_w = (const float*)d_in[15];
  const float* dt_w     = (const float*)d_in[16];
  const float* dt_b     = (const float*)d_in[17];
  const float* A_logs   = (const float*)d_in[18];
  const float* Ds       = (const float*)d_in[19];
  const float* o_ln_w   = (const float*)d_in[20];
  const float* o_ln_b   = (const float*)d_in[21];
  const float* op_w     = (const float*)d_in[22];
  const float* op_b     = (const float*)d_in[23];
  const float* v_ln2_w  = (const float*)d_in[24];
  const float* v_ln2_b  = (const float*)d_in[25];
  const float* m_fc1_w  = (const float*)d_in[26];
  const float* m_fc1_b  = (const float*)d_in[27];
  const float* m_fc2_w  = (const float*)d_in[28];
  const float* m_fc2_b  = (const float*)d_in[29];
  float* outp = (float*)d_out;

  float* ws = (float*)d_ws;
  const size_t F = (size_t)BB*CC*LL;   // 3,145,728 floats
  float* pA  = ws;          // F
  float* pB  = ws + 1*F;    // F
  float* pC  = ws + 2*F;    // 4F (expand buffer; also reused as an F-sized temp)
  float* pX1 = ws + 6*F;    // F
  float* pX2 = ws + 7*F;    // F
  float* pXD = ws + 8*F;    // B*K*8*L = 1,048,576 floats

  dim3 blk(256);
  dim3 gElem((unsigned)((F + 255)/256));
  dim3 gLN((BB*LL + 255)/256);
  dim3 gG96 (LL/128,  CC/16, BB);          // 96-output GEMMs
  dim3 gG384(LL/128, (4*CC)/16, BB);       // 384-output GEMMs

  // --- ConvNeXt-style block: x1 = x + fc2(gelu(fc1(LN(dwconv7(x))))) ---
  dwconv_kernel<7,false><<<gElem, blk, 0, stream>>>(x, cn_dw_w, cn_dw_b, pA);
  ln2d_kernel<<<gLN, blk, 0, stream>>>(pA, cn_ln_w, cn_ln_b, pB);
  gemm_wmma_kernel<1,CC  ><<<gG384, blk, 0, stream>>>(pB, cn_fc1_w, cn_fc1_b, nullptr, pC, 4*CC);
  gemm_wmma_kernel<0,4*CC><<<gG96 , blk, 0, stream>>>(pC, cn_fc2_w, cn_fc2_b, x,       pX1, CC);

  // --- SS2D: x2 = x1 + op(LN_o(cross_merge(scan(cross_scan(silu(dw3(ip(LN1(x1))))))))) ---
  ln2d_kernel<<<gLN, blk, 0, stream>>>(pX1, v_ln1_w, v_ln1_b, pB);
  gemm_wmma_kernel<0,CC><<<gG96, blk, 0, stream>>>(pB, ip_w, ip_b, nullptr, pA, CC);
  dwconv_kernel<3,true><<<gElem, blk, 0, stream>>>(pA, dw_w, dw_b, pB);          // pB = x_ss
  xdbl_kernel<<<dim3((BB*KK*LL)/256), blk, 0, stream>>>(pB, x_proj_w, pXD);
  hipMemsetAsync(pA, 0, F*sizeof(float), stream);                               // pA = y merged
  scan_kernel<<<dim3((BB*KK*CC + 127)/128), dim3(128), 0, stream>>>(
      pXD, pB, dt_w, dt_b, A_logs, Ds, pA);
  ln2d_kernel<<<gLN, blk, 0, stream>>>(pA, o_ln_w, o_ln_b, pC);
  gemm_wmma_kernel<0,CC><<<gG96, blk, 0, stream>>>(pC, op_w, op_b, pX1, pX2, CC);

  // --- MLP block: out = x2 + mfc2(gelu(mfc1(LN2(x2)))) ---
  ln2d_kernel<<<gLN, blk, 0, stream>>>(pX2, v_ln2_w, v_ln2_b, pB);
  gemm_wmma_kernel<1,CC  ><<<gG384, blk, 0, stream>>>(pB, m_fc1_w, m_fc1_b, nullptr, pC, 4*CC);
  gemm_wmma_kernel<0,4*CC><<<gG96 , blk, 0, stream>>>(pC, m_fc2_w, m_fc2_b, pX2, outp, CC);
}
